// ResNet18_37821482009119
// MI455X (gfx1250) — 8-bit matrix operand order repaired and checked by reading, compiled, not run on hardware
//
#include <hip/hip_runtime.h>
#include <cstdint>
#include <cstddef>


typedef __attribute__((ext_vector_type(8))) int v8i;
typedef unsigned long long u64;

union V8 { v8i v; int i[8]; };

#define SCALE_HT (10.0f / 127.0f)

#if defined(__has_builtin)
#if __has_builtin(__builtin_amdgcn_global_load_async_to_lds_b128) && \
    __has_builtin(__builtin_amdgcn_s_wait_asynccnt)
#define HAVE_ASYNC_LDS 1
#endif
#endif
#ifndef HAVE_ASYNC_LDS
#define HAVE_ASYNC_LDS 0
#endif

__device__ __forceinline__ float ht4(float x) {
  float c = fminf(fmaxf(x, -10.0f), 10.0f);
  float r = rintf(c / SCALE_HT);
  r = fminf(fmaxf(r, -127.0f), 127.0f);
  return r * SCALE_HT;
}

__device__ __forceinline__ uint8_t relu3_code(float x) {
  float q = rintf(fminf(fmaxf(x, 0.0f), 1.5f) * 2.0f);
  return (uint8_t)q;
}


__global__ void k_sign(const float* __restrict__ w, int8_t* __restrict__ q, size_t n) {
  size_t i = (size_t)blockIdx.x * blockDim.x + threadIdx.x;
  if (i >= n) return;
  float v = w[i];
  q[i] = (int8_t)((v > 0.0f) ? 1 : ((v < 0.0f) ? -1 : 0));
}

__global__ void k_int8_oc(const float* __restrict__ w, int8_t* __restrict__ q, int K) {
  const int oc = blockIdx.x;
  const float* row = w + (size_t)oc * K;
  __shared__ float red[256];
  float mx = 0.0f;
  for (int i = threadIdx.x; i < K; i += 256) mx = fmaxf(mx, fabsf(row[i]));
  red[threadIdx.x] = mx;
  __syncthreads();
  for (int off = 128; off > 0; off >>= 1) {
    if (threadIdx.x < off) red[threadIdx.x] = fmaxf(red[threadIdx.x], red[threadIdx.x + off]);
    __syncthreads();
  }
  float s = fmaxf(red[0] / 127.0f, 2e-16f);
  for (int i = threadIdx.x; i < K; i += 256) {
    float c = rintf(row[i] / s);
    c = fminf(fmaxf(c, -127.0f), 127.0f);
    q[(size_t)oc * K + i] = (int8_t)c;
  }
}

__global__ void k_int8_oc_f(const float* __restrict__ w, float* __restrict__ q, int K) {
  const int oc = blockIdx.x;
  const float* row = w + (size_t)oc * K;
  __shared__ float red[256];
  float mx = 0.0f;
  for (int i = threadIdx.x; i < K; i += 256) mx = fmaxf(mx, fabsf(row[i]));
  red[threadIdx.x] = mx;
  __syncthreads();
  for (int off = 128; off > 0; off >>= 1) {
    if (threadIdx.x < off) red[threadIdx.x] = fmaxf(red[threadIdx.x], red[threadIdx.x + off]);
    __syncthreads();
  }
  float s = fmaxf(red[0] / 127.0f, 2e-16f);
  for (int i = threadIdx.x; i < K; i += 256) {
    float c = rintf(row[i] / s);
    q[(size_t)oc * K + i] = fminf(fmaxf(c, -127.0f), 127.0f);
  }
}

__global__ void k_fcmax(const float* __restrict__ w, float* __restrict__ s, size_t n) {
  __shared__ float red[1024];
  float mx = 0.0f;
  for (size_t i = threadIdx.x; i < n; i += 1024) mx = fmaxf(mx, fabsf(w[i]));
  red[threadIdx.x] = mx;
  __syncthreads();
  for (int off = 512; off > 0; off >>= 1) {
    if ((int)threadIdx.x < off) red[threadIdx.x] = fmaxf(red[threadIdx.x], red[threadIdx.x + off]);
    __syncthreads();
  }
  if (threadIdx.x == 0) s[0] = fmaxf(red[0] / 127.0f, 2e-16f);
}

__global__ void k_fcq(const float* __restrict__ w, const float* __restrict__ s,
                      float* __restrict__ q, size_t n) {
  size_t i = (size_t)blockIdx.x * blockDim.x + threadIdx.x;
  if (i >= n) return;
  float c = rintf(w[i] / s[0]);
  q[i] = fminf(fmaxf(c, -127.0f), 127.0f);
}

__global__ void k_im2col(const uint8_t* __restrict__ in, uint8_t* __restrict__ out,
                         int Cin, int Hin, int Win, int Ho, int Wo,
                         int KH, int KW, int stride, int pad,
                         int NPIXin, int NPIXo) {
  const int K = Cin * KH * KW;
  size_t idx = (size_t)blockIdx.x * blockDim.x + threadIdx.x;
  size_t total = (size_t)NPIXo * K;
  if (idx >= total) return;
  int k = (int)(idx % K);
  size_t npo = idx / K;
  int wo = (int)(npo % Wo);
  size_t t = npo / Wo;
  int ho = (int)(t % Ho);
  int n = (int)(t / Ho);
  int kw = k % KW;
  int kh = (k / KW) % KH;
  int c = k / (KH * KW);
  int hi = ho * stride - pad + kh;
  int wi = wo * stride - pad + kw;
  uint8_t v = 0;
  if ((unsigned)hi < (unsigned)Hin && (unsigned)wi < (unsigned)Win)
    v = in[(size_t)c * NPIXin + ((size_t)n * Hin + hi) * Win + wi];
  out[idx] = v;
}


__device__ __forceinline__ void load_a8(V8& av, const int8_t* ap) {
  u64 q0 = *(const u64*)(ap + 0);
  u64 q1 = *(const u64*)(ap + 8);
  u64 q2 = *(const u64*)(ap + 32);
  u64 q3 = *(const u64*)(ap + 40);
  av.i[0] = (int)(unsigned)q0; av.i[1] = (int)(unsigned)(q0 >> 32);
  av.i[2] = (int)(unsigned)q1; av.i[3] = (int)(unsigned)(q1 >> 32);
  av.i[4] = (int)(unsigned)q2; av.i[5] = (int)(unsigned)(q2 >> 32);
  av.i[6] = (int)(unsigned)q3; av.i[7] = (int)(unsigned)(q3 >> 32);
}

#if HAVE_ASYNC_LDS
typedef int v4i_n __attribute__((vector_size(16)));
typedef __attribute__((address_space(1))) v4i_n* glb_v4p;
typedef __attribute__((address_space(3))) v4i_n* lds_v4p;
#define GLDS_B128(ldsp, glbp)                                                  \
  __builtin_amdgcn_global_load_async_to_lds_b128(                              \
      (glb_v4p)(uintptr_t)(const void*)(glbp),                                 \
      (lds_v4p)(uint32_t)(uintptr_t)(void*)(ldsp), 0, 0)
#endif

__global__ __launch_bounds__(128)
void k_gemm_iu8(const int8_t* __restrict__ A, const uint8_t* __restrict__ B,
                int* __restrict__ C, int OC, int K, int NPIX) {
  const int lane = threadIdx.x & 31;
  const int wave = threadIdx.x >> 5;
  const int m0 = blockIdx.y * 16;
  const int nblock = blockIdx.x * 256 + wave * 64;

  const int m = lane & 15;
  const int khalf = lane >> 4;
  const int8_t* arow = A + (size_t)(m0 + m) * K + khalf * 16;

  v8i acc0 = {}, acc1 = {}, acc2 = {}, acc3 = {};

#if HAVE_ASYNC_LDS
  __shared__ __align__(16) uint8_t sB[4][2][4096];
  uint8_t* buf0 = &sB[wave][0][0];
  uint8_t* buf1 = &sB[wave][1][0];

  size_t srcoff[8];
  uint32_t dstoff[8];
#pragma unroll
  for (int i = 0; i < 8; ++i) {
    int cidx = lane + i * 32;
    int col = cidx >> 2, piece = cidx & 3;
    int np = nblock + col;
    if (np >= NPIX) np = NPIX - 1;
    srcoff[i] = (size_t)np * K + (size_t)piece * 16;
    dstoff[i] = (uint32_t)(col * 64 + piece * 16);
  }

#pragma unroll
  for (int i = 0; i < 8; ++i) GLDS_B128(buf0 + dstoff[i], B + srcoff[i]);

  int cur = 0;
  for (int kb = 0; kb < K; kb += 64) {
    const int nkb = kb + 64;
    uint8_t* nxt = cur ? buf0 : buf1;
    if (nkb < K) {
#pragma unroll
      for (int i = 0; i < 8; ++i) GLDS_B128(nxt + dstoff[i], B + srcoff[i] + nkb);
      __builtin_amdgcn_s_wait_asynccnt(8);
    } else {
      __builtin_amdgcn_s_wait_asynccnt(0);
    }

    V8 av;
    load_a8(av, arow + kb);
    __builtin_prefetch(arow + kb + 64, 0, 3);

    const uint8_t* sb = cur ? buf1 : buf0;
#define DSB(bv, t)                                                            \
    {                                                                         \
      const int4* p = (const int4*)(sb + ((t)*16 + m) * 64 + khalf * 16);     \
      int4 lo = p[0];                                                         \
      int4 hi = p[2];                                           \
      bv.i[0] = lo.x; bv.i[1] = lo.y; bv.i[2] = lo.z; bv.i[3] = lo.w;         \
      bv.i[4] = hi.x; bv.i[5] = hi.y; bv.i[6] = hi.z; bv.i[7] = hi.w;         \
    }
    V8 b0, b1, b2, b3;
    DSB(b0, 0) DSB(b1, 1) DSB(b2, 2) DSB(b3, 3)
#undef DSB

    acc0 = __builtin_amdgcn_wmma_i32_16x16x64_iu8(true, av.v, false, b0.v, acc0, false, false);
    acc1 = __builtin_amdgcn_wmma_i32_16x16x64_iu8(true, av.v, false, b1.v, acc1, false, false);
    acc2 = __builtin_amdgcn_wmma_i32_16x16x64_iu8(true, av.v, false, b2.v, acc2, false, false);
    acc3 = __builtin_amdgcn_wmma_i32_16x16x64_iu8(true, av.v, false, b3.v, acc3, false, false);
    cur ^= 1;
  }
#else
  size_t brow0, brow1, brow2, brow3;
  {
    int np0 = nblock + 0 * 16 + m;  if (np0 >= NPIX) np0 = NPIX - 1;
    int np1 = nblock + 1 * 16 + m;  if (np1 >= NPIX) np1 = NPIX - 1;
    int np2 = nblock + 2 * 16 + m;  if (np2 >= NPIX) np2 = NPIX - 1;
    int np3 = nblock + 3 * 16 + m;  if (np3 >= NPIX) np3 = NPIX - 1;
    size_t koff = (size_t)khalf * 16;
    brow0 = (size_t)np0 * K + koff;
    brow1 = (size_t)np1 * K + koff;
    brow2 = (size_t)np2 * K + koff;
    brow3 = (size_t)np3 * K + koff;
  }

#define LOAD_B(bv, brow, kb)                                                  \
  {                                                                           \
    const uint8_t* bp = B + (brow) + (kb);                                    \
    int4 lo = *(const int4*)(bp);                                             \
    int4 hi = *(const int4*)(bp + 32);                                        \
    bv.i[0] = lo.x; bv.i[1] = lo.y; bv.i[2] = lo.z; bv.i[3] = lo.w;           \
    bv.i[4] = hi.x; bv.i[5] = hi.y; bv.i[6] = hi.z; bv.i[7] = hi.w;           \
  }

  V8 av, b0, b1, b2, b3;
  load_a8(av, arow);
  LOAD_B(b0, brow0, 0) LOAD_B(b1, brow1, 0) LOAD_B(b2, brow2, 0) LOAD_B(b3, brow3, 0)

  for (int kb = 0; kb < K; kb += 64) {
    const int nkb = kb + 64;
    V8 na, n0, n1, n2, n3;
    const bool more = nkb < K;
    if (more) {
      load_a8(na, arow + nkb);
      LOAD_B(n0, brow0, nkb) LOAD_B(n1, brow1, nkb)
      LOAD_B(n2, brow2, nkb) LOAD_B(n3, brow3, nkb)
      __builtin_prefetch(arow + nkb + 64, 0, 3);
    }
    acc0 = __builtin_amdgcn_wmma_i32_16x16x64_iu8(true, av.v, false, b0.v, acc0, false, false);
    acc1 = __builtin_amdgcn_wmma_i32_16x16x64_iu8(true, av.v, false, b1.v, acc1, false, false);
    acc2 = __builtin_amdgcn_wmma_i32_16x16x64_iu8(true, av.v, false, b2.v, acc2, false, false);
    acc3 = __builtin_amdgcn_wmma_i32_16x16x64_iu8(true, av.v, false, b3.v, acc3, false, false);
    if (more) { av = na; b0 = n0; b1 = n1; b2 = n2; b3 = n3; }
  }
#undef LOAD_B
#endif

  const int rowbase = m0 + khalf * 8;
  const int col0 = nblock + m;
  V8 d0, d1, d2, d3;
  d0.v = acc0; d1.v = acc1; d2.v = acc2; d3.v = acc3;
#define STORE_T(dv, t)                                                        \
  {                                                                           \
    int col = col0 + (t)*16;                                                  \
    if (col < NPIX) {                                                         \
      _Pragma("unroll")                                                       \
      for (int r = 0; r < 8; ++r)                                             \
        C[(size_t)(rowbase + r) * NPIX + col] = dv.i[r];                      \
    }                                                                         \
  }
  STORE_T(d0, 0) STORE_T(d1, 1) STORE_T(d2, 2) STORE_T(d3, 3)
#undef STORE_T
}

template <typename T>
__global__ void k_bnstats(const T* __restrict__ x, int NPIX,
                          float* __restrict__ mean, float* __restrict__ var) {
  const int c = blockIdx.x;
  const T* row = x + (size_t)c * NPIX;
  __shared__ float s1[256], s2[256];
  float a = 0.0f, b = 0.0f;
  for (int i = threadIdx.x; i < NPIX; i += 256) {
    float v = (float)row[i];
    a += v; b += v * v;
  }
  s1[threadIdx.x] = a; s2[threadIdx.x] = b;
  __syncthreads();
  for (int off = 128; off > 0; off >>= 1) {
    if ((int)threadIdx.x < off) {
      s1[threadIdx.x] += s1[threadIdx.x + off];
      s2[threadIdx.x] += s2[threadIdx.x + off];
    }
    __syncthreads();
  }
  if (threadIdx.x == 0) {
    float m = s1[0] / (float)NPIX;
    mean[c] = m;
    var[c] = s2[0] / (float)NPIX - m * m;
  }
}

template <typename T, int MODE>
__global__ void k_bnapply(const T* __restrict__ x, const float* __restrict__ mean,
                          const float* __restrict__ var, const float* __restrict__ g,
                          const float* __restrict__ b, int NPIX, size_t total,
                          uint8_t* __restrict__ oc8, float* __restrict__ of32) {
  size_t i = (size_t)blockIdx.x * blockDim.x + threadIdx.x;
  if (i >= total) return;
  int c = (int)(i / NPIX);
  float y = ((float)x[i] - mean[c]) * rsqrtf(var[c] + 1e-5f) * g[c] + b[c];
  if (MODE == 1) oc8[i] = relu3_code(y);
  if (MODE == 2) of32[i] = ht4(y);
}

__global__ void k_add_relu3(const float* __restrict__ a, const float* __restrict__ b,
                            uint8_t* __restrict__ out, size_t n) {
  size_t i = (size_t)blockIdx.x * blockDim.x + threadIdx.x;
  if (i >= n) return;
  out[i] = relu3_code(a[i] + b[i]);
}

__global__ void k_addx_relu3(const uint8_t* __restrict__ xcode, const float* __restrict__ o,
                             uint8_t* __restrict__ out, size_t n) {
  size_t i = (size_t)blockIdx.x * blockDim.x + threadIdx.x;
  if (i >= n) return;
  out[i] = relu3_code(ht4(0.5f * (float)xcode[i]) + o[i]);
}

__global__ void k_conv1(const float* __restrict__ x, const float* __restrict__ wq,
                        float* __restrict__ out) {
  const int NPIX = 32 * 112 * 112;
  size_t idx = (size_t)blockIdx.x * blockDim.x + threadIdx.x;
  if (idx >= (size_t)64 * NPIX) return;
  int npix = (int)(idx % NPIX);
  int oc = (int)(idx / NPIX);
  int wo = npix % 112;
  int ho = (npix / 112) % 112;
  int n = npix / (112 * 112);
  const float* wrow = wq + (size_t)oc * 147;
  float acc = 0.0f;
  for (int c = 0; c < 3; ++c) {
    for (int kh = 0; kh < 7; ++kh) {
      int hi = ho * 2 - 3 + kh;
      if ((unsigned)hi >= 224u) continue;
      const float* xr = x + (((size_t)n * 3 + c) * 224 + hi) * 224;
      const float* wr = wrow + c * 49 + kh * 7;
      for (int kw = 0; kw < 7; ++kw) {
        int wi = wo * 2 - 3 + kw;
        if ((unsigned)wi < 224u) acc = fmaf(xr[wi], wr[kw], acc);
      }
    }
  }
  out[idx] = acc;
}

__global__ void k_maxpool(const float* __restrict__ in, float* __restrict__ out) {
  const int NPIXo = 32 * 56 * 56;
  size_t idx = (size_t)blockIdx.x * blockDim.x + threadIdx.x;
  if (idx >= (size_t)64 * NPIXo) return;
  int npix = (int)(idx % NPIXo);
  int c = (int)(idx / NPIXo);
  int wo = npix % 56;
  int ho = (npix / 56) % 56;
  int n = npix / (56 * 56);
  const float* base = in + ((size_t)c * 32 + n) * 112 * 112;
  float mx = -INFINITY;
  for (int kh = 0; kh < 3; ++kh) {
    int hi = ho * 2 - 1 + kh;
    if ((unsigned)hi >= 112u) continue;
    for (int kw = 0; kw < 3; ++kw) {
      int wi = wo * 2 - 1 + kw;
      if ((unsigned)wi < 112u) mx = fmaxf(mx, base[(size_t)hi * 112 + wi]);
    }
  }
  out[idx] = mx;
}

__global__ void k_avgpool(const uint8_t* __restrict__ codes, float* __restrict__ pooled) {
  const int NPIX = 32 * 49;
  int idx = blockIdx.x * blockDim.x + threadIdx.x;
  if (idx >= 32 * 512) return;
  int b = idx / 512, c = idx % 512;
  int s = 0;
  const uint8_t* p = codes + (size_t)c * NPIX + b * 49;
  for (int j = 0; j < 49; ++j) s += p[j];
  pooled[idx] = (float)s * 0.5f / 49.0f;
}

__global__ void k_fc(const float* __restrict__ pooled, const float* __restrict__ wq,
                     const float* __restrict__ s, float* __restrict__ out) {
  int idx = blockIdx.x * blockDim.x + threadIdx.x;
  if (idx >= 32 * 1000) return;
  int b = idx / 1000, j = idx % 1000;
  const float* xr = pooled + (size_t)b * 512;
  const float* wr = wq + (size_t)j * 512;
  float acc = 0.0f;
  for (int c = 0; c < 512; ++c) acc = fmaf(xr[c], wr[c], acc);
  out[idx] = acc * s[0];
}

static inline unsigned gblocks(size_t n, int t) { return (unsigned)((n + t - 1) / t); }

struct Scratch {
  uint8_t* i2c; int8_t* wq; int* ga; float* mean; float* var;
};

static void conv_bn_q(hipStream_t stream, const Scratch& S,
                      const uint8_t* in_codes, int Cin, int Hin,
                      int KH, int stride, int pad,
                      const float* w, int wmode  ,
                      const float* g, const float* b,
                      int outmode  ,
                      uint8_t* out_codes, float* out_f32, int OC) {
  const int Ho = (Hin + 2 * pad - KH) / stride + 1;
  const int K = Cin * KH * KH;
  const int NPIXin = 32 * Hin * Hin;
  const int NPIXo = 32 * Ho * Ho;

  if (wmode == 0)
    k_sign<<<gblocks((size_t)OC * K, 256), 256, 0, stream>>>(w, S.wq, (size_t)OC * K);
  else
    k_int8_oc<<<OC, 256, 0, stream>>>(w, S.wq, K);

  size_t tot = (size_t)NPIXo * K;
  k_im2col<<<gblocks(tot, 256), 256, 0, stream>>>(in_codes, S.i2c, Cin, Hin, Hin,
                                                  Ho, Ho, KH, KH, stride, pad,
                                                  NPIXin, NPIXo);

  dim3 gg(gblocks((size_t)NPIXo, 256), OC / 16);
  k_gemm_iu8<<<gg, 128, 0, stream>>>(S.wq, S.i2c, S.ga, OC, K, NPIXo);

  k_bnstats<int><<<OC, 256, 0, stream>>>(S.ga, NPIXo, S.mean, S.var);

  size_t n = (size_t)OC * NPIXo;
  if (outmode == 1)
    k_bnapply<int, 1><<<gblocks(n, 256), 256, 0, stream>>>(S.ga, S.mean, S.var, g, b,
                                                           NPIXo, n, out_codes, nullptr);
  else
    k_bnapply<int, 2><<<gblocks(n, 256), 256, 0, stream>>>(S.ga, S.mean, S.var, g, b,
                                                           NPIXo, n, nullptr, out_f32);
}

extern "C" void kernel_launch(void* const* d_in, const int* in_sizes, int n_in,
                              void* d_out, int out_size, void* d_ws, size_t ws_size,
                              hipStream_t stream) {
  (void)in_sizes; (void)n_in; (void)out_size; (void)ws_size;

  const float* X = (const float*)d_in[0];
  const float* conv1_w = (const float*)d_in[1];
  const float* bn1_g = (const float*)d_in[2];
  const float* bn1_b = (const float*)d_in[3];
  const float* fc_w = (const float*)d_in[64];

  char* ws = (char*)d_ws;
  size_t off = 0;
  auto alloc = [&](size_t bytes) -> void* {
    void* p = ws + off;
    off = (off + bytes + 255) & ~(size_t)255;
    return p;
  };
  float* F0 = (float*)alloc(102760448);
  float* F1 = (float*)alloc(25690112);
  uint8_t* I2C = (uint8_t*)alloc(57802752);
  int* GA = (int*)alloc(25690112);
  uint8_t* C0 = (uint8_t*)alloc(6422528);
  uint8_t* C1 = (uint8_t*)alloc(6422528);
  uint8_t* C2 = (uint8_t*)alloc(6422528);
  int8_t* WQ = (int8_t*)alloc(2359296);
  float* MEAN = (float*)alloc(512 * 4);
  float* VAR = (float*)alloc(512 * 4);
  float* WQF = (float*)alloc(64 * 147 * 4);
  float* POOL = (float*)alloc(32 * 512 * 4);
  float* FCW = (float*)alloc(512 * 1000 * 4);
  float* FCS = (float*)alloc(256);

  Scratch S{I2C, WQ, GA, MEAN, VAR};

  k_int8_oc_f<<<64, 256, 0, stream>>>(conv1_w, WQF, 147);
  {
    size_t n1 = (size_t)64 * 32 * 112 * 112;
    k_conv1<<<gblocks(n1, 256), 256, 0, stream>>>(X, WQF, F0);
    size_t n2 = (size_t)64 * 32 * 56 * 56;
    k_maxpool<<<gblocks(n2, 256), 256, 0, stream>>>(F0, F1);
    k_bnstats<float><<<64, 256, 0, stream>>>(F1, 32 * 56 * 56, MEAN, VAR);
    k_bnapply<float, 1><<<gblocks(n2, 256), 256, 0, stream>>>(
        F1, MEAN, VAR, bn1_g, bn1_b, 32 * 56 * 56, n2, C0, nullptr);
  }

  static const int COUT[4] = {64, 128, 256, 512};
  static const int S0A[4] = {1, 2, 2, 2};
  int H = 56, Cin = 64;
  for (int i = 0; i < 4; ++i) {
    const int base = 4 + i * 15;
    const int cout = COUT[i];
    const int s0 = S0A[i];
    const int Ho = (H - 1) / s0 + 1;
    const size_t n = (size_t)cout * 32 * Ho * Ho;

    conv_bn_q(stream, S, C0, Cin, H, 1, s0, 0, (const float*)d_in[base + 2], 1,
              (const float*)d_in[base + 7], (const float*)d_in[base + 8], 2,
              nullptr, F0, cout);
    conv_bn_q(stream, S, C0, Cin, H, 3, s0, 1, (const float*)d_in[base + 0], 0,
              (const float*)d_in[base + 3], (const float*)d_in[base + 4], 1,
              C1, nullptr, cout);
    conv_bn_q(stream, S, C1, cout, Ho, 3, 1, 1, (const float*)d_in[base + 1], 0,
              (const float*)d_in[base + 5], (const float*)d_in[base + 6], 2,
              nullptr, F1, cout);
    k_add_relu3<<<gblocks(n, 256), 256, 0, stream>>>(F0, F1, C2, n);

    conv_bn_q(stream, S, C2, cout, Ho, 3, 1, 1, (const float*)d_in[base + 9], 0,
              (const float*)d_in[base + 11], (const float*)d_in[base + 12], 1,
              C1, nullptr, cout);
    conv_bn_q(stream, S, C1, cout, Ho, 3, 1, 1, (const float*)d_in[base + 10], 0,
              (const float*)d_in[base + 13], (const float*)d_in[base + 14], 2,
              nullptr, F1, cout);
    k_addx_relu3<<<gblocks(n, 256), 256, 0, stream>>>(C2, F1, C0, n);

    H = Ho;
    Cin = cout;
  }

  k_fcmax<<<1, 1024, 0, stream>>>(fc_w, FCS, (size_t)512 * 1000);
  k_fcq<<<gblocks((size_t)512 * 1000, 256), 256, 0, stream>>>(fc_w, FCS, FCW,
                                                              (size_t)512 * 1000);
  k_avgpool<<<gblocks((size_t)32 * 512, 256), 256, 0, stream>>>(C0, POOL);
  k_fc<<<gblocks((size_t)32 * 1000, 256), 256, 0, stream>>>(POOL, FCW, FCS,
                                                            (float*)d_out);
}
